// PT2_VoxelBackBone8x_UniBN_27419071218044
// MI455X (gfx1250) — matrix operand order repaired and checked by reading, not compiled, not run on hardware
//
#include <hip/hip_runtime.h>
#include <hip/hip_bf16.h>

typedef _Float16 f16;
typedef __attribute__((ext_vector_type(16))) _Float16 v16h;
typedef __attribute__((ext_vector_type(8)))  _Float16 v8h;
typedef __attribute__((ext_vector_type(4)))  _Float16 v4h;
typedef __attribute__((ext_vector_type(8)))  float    v8f;

// ---------------- mask kernels ----------------
__global__ void k_mask_init(const float* __restrict__ occ, float* __restrict__ m, int n) {
  int i = blockIdx.x * blockDim.x + threadIdx.x;
  if (i < n) m[i] = (occ[i] < 0.05f) ? 1.f : 0.f;
}

__global__ void k_dilate(const float* __restrict__ mi, float* __restrict__ mo,
                         int Di, int Hi, int Wi, int Do, int Ho, int Wo,
                         int kd, int kh, int kw, int sd, int sh, int sw,
                         int pd, int ph, int pw) {
  int i = blockIdx.x * blockDim.x + threadIdx.x;
  int n = Do * Ho * Wo;
  if (i >= n) return;
  int w = i % Wo, h = (i / Wo) % Ho, d = i / (Wo * Ho);
  float acc = 0.f;
  for (int td = 0; td < kd; ++td) {
    int z = d * sd + td - pd; if (z < 0 || z >= Di) continue;
    for (int th = 0; th < kh; ++th) {
      int y = h * sh + th - ph; if (y < 0 || y >= Hi) continue;
      for (int tw = 0; tw < kw; ++tw) {
        int x = w * sw + tw - pw; if (x < 0 || x >= Wi) continue;
        acc += mi[(z * Hi + y) * Wi + x];
      }
    }
  }
  mo[i] = (acc > 0.f) ? 1.f : 0.f;
}

__global__ void k_masksum(const float* __restrict__ m, int n, float* __restrict__ cnt) {
  __shared__ float s[256];
  float a = 0.f;
  for (int i = blockIdx.x * blockDim.x + threadIdx.x; i < n; i += gridDim.x * blockDim.x)
    a += m[i];
  s[threadIdx.x] = a; __syncthreads();
  for (int o = 128; o > 0; o >>= 1) {
    if ((int)threadIdx.x < o) s[threadIdx.x] += s[threadIdx.x + o];
    __syncthreads();
  }
  if (threadIdx.x == 0) atomicAdd(cnt, s[0]);
}

// ---------------- activation init: x (planar f32) * m1 -> channel-last f16 [vox][4]
__global__ void k_init_act(const float* __restrict__ x, const float* __restrict__ m,
                           f16* __restrict__ act, int nvox) {
  int v = blockIdx.x * blockDim.x + threadIdx.x;
  if (v >= nvox) return;
  float mv = m[v];
  v4h o;
  #pragma unroll
  for (int c = 0; c < 4; ++c) o[c] = (f16)(x[c * nvox + v] * mv);
  *(v4h*)(act + (size_t)v * 4) = o;
}

// ---------------- weight pack: f32 [Co][Ci][taps] -> f16 B tiles [coTile][KC][16][32]
__global__ void k_pack_w(const float* __restrict__ w, f16* __restrict__ bp,
                         int Cin, int lc, int ntaps, int KC, int total) {
  int cmask = Cin - 1;
  for (int i = blockIdx.x * blockDim.x + threadIdx.x; i < total; i += gridDim.x * blockDim.x) {
    int k = i & 31, nrow = (i >> 5) & 15, t = i >> 9;
    int cot = t / KC, kc = t - cot * KC;
    int kk = kc * 32 + k;
    int tap = kk >> lc, c = kk & cmask;
    int co = cot * 16 + nrow;
    float v = 0.f;
    if (tap < ntaps) v = w[(co * Cin + c) * ntaps + tap];
    bp[i] = (f16)v;
  }
}

// ---------------- implicit-GEMM conv via WMMA f32<-f16, async-DMA A pack ---------
// act: channel-last f16 [Di*Hi*Wi][Cin].  y: planar f16 [Cout][Do*Ho*Wo].
// block (32, Cout/16): one 16-voxel M tile along W, one 16-cout tile per wave.
// A tiles stream into LDS via GLOBAL_LOAD_ASYNC_TO_LDS (ASYNCcnt), double-buffered,
// so the chunk-(k+1) DMA overlaps the chunk-k WMMA. OOB taps read a zeroed pad.
template <int LC, int KD, int KH, int KW>
__global__ void __launch_bounds__(256) k_conv(
    const f16* __restrict__ act, const f16* __restrict__ bp, f16* __restrict__ y,
    const f16* __restrict__ zeropad,
    int Di, int Hi, int Wi, int Do, int Ho, int Wo,
    int sd, int sh, int sw, int pd, int ph, int pw) {
  constexpr int CIN = 1 << LC;
  constexpr int NTAPS = KD * KH * KW;
  constexpr int KHW = KH * KW;
  constexpr int KC = (NTAPS * CIN + 31) / 32;
  constexpr int U = (CIN >= 8) ? 8 : 4;   // pack unit (halves), contiguous within a tap
  constexpr int UPR = 32 / U;             // units per A-row
  constexpr int NUNITS = 512 / U;

  __shared__ __align__(16) f16 As[2][16 * 32];
  const int lane = threadIdx.x;
  const int wv = threadIdx.y;
  const int nthreads = blockDim.x * blockDim.y;
  const int tid = wv * 32 + lane;
  const int w0 = blockIdx.x * 16;
  const int h = blockIdx.y;
  const int d = blockIdx.z;
  const int row = lane & 15, sel = lane >> 4;
  const int z0 = d * sd - pd, y0 = h * sh - ph;

  // issue async global->LDS copies for K-chunk kc into buffer asb
  auto issue = [&](int kc, f16* asb) {
    for (int j = tid; j < NUNITS; j += nthreads) {
      int m = j / UPR, ku = (j % UPR) * U;
      int kk = kc * 32 + ku;
      int tap = kk >> LC, c0 = kk & (CIN - 1);
      int td = tap / KHW, r = tap - td * KHW;
      int th = r / KW, tw = r - th * KW;
      int z = z0 + td, yy = y0 + th, x = (w0 + m) * sw + tw - pw;
      bool ok = (tap < NTAPS) && z >= 0 && z < Di && yy >= 0 && yy < Hi && x >= 0 && x < Wi;
      const f16* src = ok ? act + (size_t)((z * Hi + yy) * Wi + x) * CIN + c0 : zeropad;
      unsigned ldsaddr = (unsigned)(size_t)(asb + m * 32 + ku);
      if constexpr (U == 8) {
        asm volatile("global_load_async_to_lds_b128 %0, %1, off"
                     :: "v"(ldsaddr), "v"(src) : "memory");
      } else {
        asm volatile("global_load_async_to_lds_b64 %0, %1, off"
                     :: "v"(ldsaddr), "v"(src) : "memory");
      }
    }
  };

  v8f acc = {};
  issue(0, As[0]);
  for (int kc = 0; kc < KC; ++kc) {
    f16* asb = As[kc & 1];
    asm volatile("s_wait_asynccnt 0x0" ::: "memory");  // own chunk-kc copies landed
    __syncthreads();                                   // all waves' copies visible
    // A fragment: lane<16 -> row=lane, K {0..7,16..23}; lane>=16 -> K {8..15,24..31}
    const f16* ar = asb + row * 32 + (sel ? 8 : 0);
    v8h a0 = *(const v8h*)ar;
    v8h a1 = *(const v8h*)(ar + 16);
    v16h af = __builtin_shufflevector(a0, a1, 0,1,2,3,4,5,6,7,8,9,10,11,12,13,14,15);
    // B fragment: lane<16 -> col N=row, K {0..7,16..23}; lane>=16 -> K {8..15,24..31}
    const f16* br = bp + (((wv * KC + kc) * 16 + row) * 32 + (sel ? 8 : 0));
    v8h b0 = *(const v8h*)br;
    v8h b1 = *(const v8h*)(br + 16);
    v16h bf = __builtin_shufflevector(b0, b1, 0,1,2,3,4,5,6,7,8,9,10,11,12,13,14,15);
    // prefetch next chunk via async DMA; overlaps the WMMA below
    if (kc + 1 < KC) issue(kc + 1, As[(kc + 1) & 1]);
    acc = __builtin_amdgcn_wmma_f32_16x16x32_f16(false, af, false, bf, (short)0, acc,
                                                 false, false);
  }
  // D layout: VGPR v, lane<16 -> (M=v, N=lane); lane>=16 -> (M=v+8, N=lane-16)
  int co = wv * 16 + row;
  int mbase = sel * 8;
  int plane = (co * Do + d) * Ho * Wo + h * Wo;
  v8h o;
  #pragma unroll
  for (int v = 0; v < 8; ++v) o[v] = (f16)acc[v];
  if (w0 + 16 <= Wo) {
    f16* dst = y + plane + w0 + mbase;                 // 8B-aligned for all levels
    *(v4h*)dst       = __builtin_shufflevector(o, o, 0, 1, 2, 3);
    *(v4h*)(dst + 4) = __builtin_shufflevector(o, o, 4, 5, 6, 7);
  } else {
    for (int v = 0; v < 8; ++v) {
      int w = w0 + mbase + v;
      if (w < Wo) y[plane + w] = o[v];
    }
  }
}

// ---------------- masked per-channel stats (sum, sumsq) on planar y ----------------
__global__ void k_stats(const f16* __restrict__ y, const float* __restrict__ m,
                        int nvox, float* __restrict__ stats) {
  __shared__ float ss[256], sq[256];
  int c = blockIdx.y;
  const f16* yc = y + (size_t)c * nvox;
  float s = 0.f, q = 0.f;
  for (int i = blockIdx.x * blockDim.x + threadIdx.x; i < nvox; i += gridDim.x * blockDim.x) {
    float v = (float)yc[i] * m[i];
    s += v; q += v * v;
  }
  ss[threadIdx.x] = s; sq[threadIdx.x] = q; __syncthreads();
  for (int o = 128; o > 0; o >>= 1) {
    if ((int)threadIdx.x < o) { ss[threadIdx.x] += ss[threadIdx.x + o]; sq[threadIdx.x] += sq[threadIdx.x + o]; }
    __syncthreads();
  }
  if (threadIdx.x == 0) { atomicAdd(&stats[2 * c], ss[0]); atomicAdd(&stats[2 * c + 1], sq[0]); }
}

__global__ void k_finalize(const float* __restrict__ stats, const float* __restrict__ cnt,
                           const float* __restrict__ g, const float* __restrict__ b,
                           float* __restrict__ sb, int Cout) {
  int c = threadIdx.x;
  if (c < Cout) {
    float n = fmaxf(*cnt, 1.f);
    float mean = stats[2 * c] / n;
    float var = fmaxf(stats[2 * c + 1] / n - mean * mean, 0.f);
    float inv = rsqrtf(var + 1e-3f);
    float sc = g[c] * inv;
    sb[c] = sc;
    sb[128 + c] = b[c] - mean * sc;
  }
}

// norm+relu+mask -> channel-last f16 activations
__global__ void k_apply_cl(const f16* __restrict__ y, const float* __restrict__ m,
                           const float* __restrict__ sb, f16* __restrict__ act,
                           int Cout, int nvox) {
  int v = blockIdx.x * blockDim.x + threadIdx.x;
  if (v >= nvox) return;
  float mv = m[v];
  f16* dst = act + (size_t)v * Cout;
  for (int c0 = 0; c0 < Cout; c0 += 8) {
    v8h o;
    #pragma unroll
    for (int j = 0; j < 8; ++j) {
      int c = c0 + j;
      float val = (float)y[(size_t)c * nvox + v] * sb[c] + sb[128 + c];
      o[j] = (f16)(fmaxf(val, 0.f) * mv);
    }
    *(v8h*)(dst + c0) = o;
  }
}

// last layer: norm+relu+mask -> planar f32 output
__global__ void k_apply_out(const f16* __restrict__ y, const float* __restrict__ m,
                            const float* __restrict__ sb, float* __restrict__ fout,
                            int nvox, int total) {
  int i = blockIdx.x * blockDim.x + threadIdx.x;
  if (i >= total) return;
  int c = i / nvox, v = i - c * nvox;
  float val = (float)y[i] * sb[c] + sb[128 + c];
  fout[i] = fmaxf(val, 0.f) * m[v];
}

// ---------------- host orchestration ----------------
struct Cfg { int Cin, Cout, kd, kh, kw, sd, sh, sw, pd, ph, pw, li, lo; };

static const int LD[5] = {41, 21, 11, 5, 2};
static const int LH[5] = {160, 80, 40, 20, 20};
static const int LW[5] = {160, 80, 40, 20, 20};
static const int LN[5] = {41 * 160 * 160, 21 * 80 * 80, 11 * 40 * 40, 5 * 20 * 20, 2 * 20 * 20};

static const Cfg CFGS[12] = {
  { 4,  16, 3,3,3, 1,1,1, 1,1,1, 0,0},  // conv_input (subm)
  {16,  16, 3,3,3, 1,1,1, 1,1,1, 0,0},  // conv1
  {16,  32, 3,3,3, 2,2,2, 1,1,1, 0,1},  // conv2_1 (strided)
  {32,  32, 3,3,3, 1,1,1, 1,1,1, 1,1},  // conv2_2
  {32,  32, 3,3,3, 1,1,1, 1,1,1, 1,1},  // conv2_3
  {32,  64, 3,3,3, 2,2,2, 1,1,1, 1,2},  // conv3_1
  {64,  64, 3,3,3, 1,1,1, 1,1,1, 2,2},  // conv3_2
  {64,  64, 3,3,3, 1,1,1, 1,1,1, 2,2},  // conv3_3
  {64,  64, 3,3,3, 2,2,2, 0,1,1, 2,3},  // conv4_1 (pad 0,1,1)
  {64,  64, 3,3,3, 1,1,1, 1,1,1, 3,3},  // conv4_2
  {64,  64, 3,3,3, 1,1,1, 1,1,1, 3,3},  // conv4_3
  {64, 128, 3,1,1, 2,1,1, 0,0,0, 3,4},  // conv_out
};

template <int LC, int KD, int KH, int KW>
static void launch_conv(const Cfg& c, const f16* act, const f16* bp, f16* y,
                        const f16* zeropad, hipStream_t stream) {
  int Di = LD[c.li], Hi = LH[c.li], Wi = LW[c.li];
  int Do = LD[c.lo], Ho = LH[c.lo], Wo = LW[c.lo];
  dim3 grid((Wo + 15) / 16, Ho, Do), block(32, c.Cout / 16);
  k_conv<LC, KD, KH, KW><<<grid, block, 0, stream>>>(act, bp, y, zeropad,
      Di, Hi, Wi, Do, Ho, Wo, c.sd, c.sh, c.sw, c.pd, c.ph, c.pw);
}

extern "C" void kernel_launch(void* const* d_in, const int* in_sizes, int n_in,
                              void* d_out, int out_size, void* d_ws, size_t ws_size,
                              hipStream_t stream) {
  (void)in_sizes; (void)n_in; (void)out_size; (void)ws_size;
  const float* x   = (const float*)d_in[0];
  const float* occ = (const float*)d_in[1];

  char* ws = (char*)d_ws;
  size_t off = 0;
  auto alloc = [&](size_t bytes) -> void* {
    void* p = ws + off; off += (bytes + 255) & ~(size_t)255; return p;
  };
  const size_t maxPlane = (size_t)16 * LN[0];             // 16 ch @ full res (max C*vox)
  f16* act  = (f16*)alloc(maxPlane * sizeof(f16));        // channel-last activations
  f16* ybuf = (f16*)alloc(maxPlane * sizeof(f16));        // planar conv outputs
  float* mk[5];
  for (int l = 0; l < 5; ++l) mk[l] = (float*)alloc((size_t)LN[l] * sizeof(float));
  float* cnt   = (float*)alloc(5 * sizeof(float));
  float* stats = (float*)alloc(256 * sizeof(float));
  float* sb    = (float*)alloc(256 * sizeof(float));
  f16* zeropad = (f16*)alloc(256);                        // zero source for OOB DMA
  f16* bpk[12]; int KCs[12], LCs[12];
  for (int i = 0; i < 12; ++i) {
    const Cfg& c = CFGS[i];
    int ntaps = c.kd * c.kh * c.kw;
    int KC = (ntaps * c.Cin + 31) / 32;
    KCs[i] = KC;
    LCs[i] = (c.Cin == 4) ? 2 : (c.Cin == 16) ? 4 : (c.Cin == 32) ? 5 : 6;
    bpk[i] = (f16*)alloc((size_t)(c.Cout / 16) * KC * 512 * sizeof(f16));
  }

  hipMemsetAsync(zeropad, 0, 256, stream);

  // masks + counts
  k_mask_init<<<(LN[0] + 255) / 256, 256, 0, stream>>>(occ, mk[0], LN[0]);
  k_dilate<<<(LN[1] + 255) / 256, 256, 0, stream>>>(mk[0], mk[1], LD[0], LH[0], LW[0],
      LD[1], LH[1], LW[1], 3, 3, 3, 2, 2, 2, 1, 1, 1);
  k_dilate<<<(LN[2] + 255) / 256, 256, 0, stream>>>(mk[1], mk[2], LD[1], LH[1], LW[1],
      LD[2], LH[2], LW[2], 3, 3, 3, 2, 2, 2, 1, 1, 1);
  k_dilate<<<(LN[3] + 255) / 256, 256, 0, stream>>>(mk[2], mk[3], LD[2], LH[2], LW[2],
      LD[3], LH[3], LW[3], 3, 3, 3, 2, 2, 2, 0, 1, 1);
  k_dilate<<<(LN[4] + 255) / 256, 256, 0, stream>>>(mk[3], mk[4], LD[3], LH[3], LW[3],
      LD[4], LH[4], LW[4], 3, 1, 1, 2, 1, 1, 0, 0, 0);
  hipMemsetAsync(cnt, 0, 5 * sizeof(float), stream);
  for (int l = 0; l < 5; ++l)
    k_masksum<<<64, 256, 0, stream>>>(mk[l], LN[l], cnt + l);

  // initial activations h0 = x * m1 -> channel-last f16
  k_init_act<<<(LN[0] + 255) / 256, 256, 0, stream>>>(x, mk[0], act, LN[0]);

  // pre-pack all conv weights into WMMA-B layout
  for (int i = 0; i < 12; ++i) {
    const Cfg& c = CFGS[i];
    int ntaps = c.kd * c.kh * c.kw;
    int total = (c.Cout / 16) * KCs[i] * 512;
    k_pack_w<<<(total + 255) / 256, 256, 0, stream>>>(
        (const float*)d_in[2 + 3 * i], bpk[i], c.Cin, LCs[i], ntaps, KCs[i], total);
  }

  // layers
  for (int i = 0; i < 12; ++i) {
    const Cfg& c = CFGS[i];
    const float* g = (const float*)d_in[3 + 3 * i];
    const float* b = (const float*)d_in[4 + 3 * i];
    int nvox = LN[c.lo];

    switch (i) {
      case 0:  launch_conv<2, 3, 3, 3>(c, act, bpk[i], ybuf, zeropad, stream); break;
      case 1:
      case 2:  launch_conv<4, 3, 3, 3>(c, act, bpk[i], ybuf, zeropad, stream); break;
      case 3:
      case 4:
      case 5:  launch_conv<5, 3, 3, 3>(c, act, bpk[i], ybuf, zeropad, stream); break;
      case 11: launch_conv<6, 3, 1, 1>(c, act, bpk[i], ybuf, zeropad, stream); break;
      default: launch_conv<6, 3, 3, 3>(c, act, bpk[i], ybuf, zeropad, stream); break;
    }

    hipMemsetAsync(stats, 0, 2 * c.Cout * sizeof(float), stream);
    dim3 sgrid(48, c.Cout);
    k_stats<<<sgrid, 256, 0, stream>>>(ybuf, mk[c.lo], nvox, stats);
    k_finalize<<<1, 128, 0, stream>>>(stats, cnt + c.lo, g, b, sb, c.Cout);

    if (i == 11) {
      int tot = c.Cout * nvox;
      k_apply_out<<<(tot + 255) / 256, 256, 0, stream>>>(ybuf, mk[c.lo], sb,
          (float*)d_out, nvox, tot);
    } else {
      k_apply_cl<<<(nvox + 255) / 256, 256, 0, stream>>>(ybuf, mk[c.lo], sb, act,
          c.Cout, nvox);
    }
  }
}
